// FoldNetEncoder_11209864643061
// MI455X (gfx1250) — matrix operand order repaired and checked by reading, not compiled, not run on hardware
//
#include <hip/hip_runtime.h>
#include <hip/hip_bf16.h>

typedef __attribute__((ext_vector_type(16))) _Float16 v16h;
typedef __attribute__((ext_vector_type(8)))  _Float16 v8h;
typedef __attribute__((ext_vector_type(8)))  float    v8f;

#define BATCH 16
#define NPTS  2048
#define KNN   16
#define BN    (BATCH * NPTS)

// ---------------------------------------------------------------------------
// KNN: one thread per query point, LDS-tiled candidates, sorted top-16
// (descending neg-dist => nearest first; strict compares keep lower index on
// ties, matching jax.lax.top_k). Self (dist 0) lands in slot 0.
// ---------------------------------------------------------------------------
__global__ __launch_bounds__(256) void knn_k(const float* __restrict__ pts,
                                             int* __restrict__ idx) {
  __shared__ float sx[256], sy[256], sz[256], ss[256];
  int b = blockIdx.x >> 3;
  int n = ((blockIdx.x & 7) << 8) + threadIdx.x;
  const float* pb = pts + (size_t)b * NPTS * 3;
  float qx = pb[n * 3 + 0], qy = pb[n * 3 + 1], qz = pb[n * 3 + 2];
  float qq = qx * qx + qy * qy + qz * qz;
  float dist[KNN];
  int   ida[KNN];
#pragma unroll
  for (int j = 0; j < KNN; ++j) { dist[j] = -3.4e38f; ida[j] = 0; }
  for (int t = 0; t < NPTS / 256; ++t) {
    int ti = t * 256 + threadIdx.x;
    float x = pb[ti * 3 + 0], y = pb[ti * 3 + 1], z = pb[ti * 3 + 2];
    __syncthreads();
    sx[threadIdx.x] = x; sy[threadIdx.x] = y; sz[threadIdx.x] = z;
    ss[threadIdx.x] = x * x + y * y + z * z;
    __syncthreads();
    for (int j = 0; j < 256; ++j) {
      float nd = 2.f * (qx * sx[j] + qy * sy[j] + qz * sz[j]) - qq - ss[j];
      if (nd > dist[KNN - 1]) {
        int p = KNN - 1;
        while (p > 0 && dist[p - 1] < nd) {
          dist[p] = dist[p - 1]; ida[p] = ida[p - 1]; --p;
        }
        dist[p] = nd; ida[p] = t * 256 + j;
      }
    }
  }
  int* op = idx + ((size_t)b * NPTS + n) * KNN;
#pragma unroll
  for (int j = 0; j < KNN; ++j) op[j] = ida[j];
}

// ---------------------------------------------------------------------------
// local_cov features: [xyz(3), outer(nbr0, nbr1)(9)], zero-padded to K=32 f16
// ---------------------------------------------------------------------------
__global__ __launch_bounds__(256) void cov_k(const float* __restrict__ pts,
                                             const int* __restrict__ idx,
                                             _Float16* __restrict__ A0) {
  int m = blockIdx.x * 256 + threadIdx.x;
  int b = m >> 11, n = m & (NPTS - 1);
  const float* pb = pts + (size_t)b * NPTS * 3;
  int i0 = idx[(size_t)m * KNN + 0];
  int i1 = idx[(size_t)m * KNN + 1];
  float p0[3], a0[3], a1[3];
#pragma unroll
  for (int c = 0; c < 3; ++c) {
    p0[c] = pb[n * 3 + c];
    a0[c] = pb[i0 * 3 + c];
    a1[c] = pb[i1 * 3 + c];
  }
  _Float16* row = A0 + (size_t)m * 32;
#pragma unroll
  for (int c = 0; c < 3; ++c) row[c] = (_Float16)p0[c];
#pragma unroll
  for (int i = 0; i < 3; ++i)
#pragma unroll
    for (int j = 0; j < 3; ++j)
      row[3 + i * 3 + j] = (_Float16)(a0[i] * a1[j]);
#pragma unroll
  for (int c = 12; c < 32; ++c) row[c] = (_Float16)0.f;
}

// ---------------------------------------------------------------------------
// Weight convert f32 -> f16 with K padding (zero fill)
// ---------------------------------------------------------------------------
__global__ void wcvt_k(const float* __restrict__ src, _Float16* __restrict__ dst,
                       int O, int Ks, int Kp) {
  int i = blockIdx.x * 256 + threadIdx.x;
  if (i >= O * Kp) return;
  int o = i / Kp, k = i - o * Kp;
  dst[i] = (k < Ks) ? (_Float16)src[o * Ks + k] : (_Float16)0.f;
}

__device__ inline void atomicMaxFloat(float* addr, float val) {
  if (val >= 0.f) atomicMax((int*)addr, __float_as_int(val));
  else            atomicMin((unsigned int*)addr, __float_as_uint(val));
}

// ---------------------------------------------------------------------------
// WMMA GEMM: out[m,o] = act( A[m,:] . W[o,:] + bias[o] )
// A: (Mtot,K) f16 row-major, W: (O,K) f16 row-major, K % 32 == 0, O % 64 == 0.
// One wave computes a 16x64 strip: A fragment loaded once per K-tile and
// reused across 4 WMMAs (4x lower A traffic, 4 wmma per 5 fragment loads).
// mode: 0 = f16 store, 1 = f32 store,
//       2 = atomic-max into outF[(m>>11)*O + o] (fused global max over pts).
// Fragment layouts used here (A/B: contiguous k, same order in both):
//   A lane l: row m=l&15; halves j   -> K = kt+(l>>4)*16+j
//   B lane l: col o=l&15; halves j   -> K = kt+(l>>4)*16+j  (contig. W row)
//   D lane l: col o=l&15; vgpr r     -> row m = (l>>4)*8 + r
// ---------------------------------------------------------------------------
__global__ __launch_bounds__(32) void gemm_k(const _Float16* __restrict__ A,
                                             const _Float16* __restrict__ W,
                                             const float* __restrict__ bias,
                                             _Float16* __restrict__ outH,
                                             float* __restrict__ outF,
                                             int K, int O, int relu, int mode) {
  int m0 = blockIdx.x * 16, o0 = blockIdx.y * 64;
  int lane = threadIdx.x;
  int lo = lane & 15, hi = lane >> 4;
  const _Float16* arow = A + (size_t)(m0 + lo) * K;
  const _Float16* wrow = W + (size_t)(o0 + lo) * K + hi * 16;
  size_t wstep = (size_t)16 * K;  // 16 output channels ahead
  v8f c0 = {}, c1 = {}, c2 = {}, c3 = {};
  for (int kt = 0; kt < K; kt += 32) {
    v8h al = *(const v8h*)(arow + kt + hi * 16);
    v8h ah = *(const v8h*)(arow + kt + hi * 16 + 8);
    v16h a;
#pragma unroll
    for (int j = 0; j < 8; ++j) { a[j] = al[j]; a[j + 8] = ah[j]; }
    v16h b0 = *(const v16h*)(wrow + kt);
    v16h b1 = *(const v16h*)(wrow + wstep + kt);
    v16h b2 = *(const v16h*)(wrow + 2 * wstep + kt);
    v16h b3 = *(const v16h*)(wrow + 3 * wstep + kt);
    c0 = __builtin_amdgcn_wmma_f32_16x16x32_f16(false, a, false, b0, (short)0, c0, false, false);
    c1 = __builtin_amdgcn_wmma_f32_16x16x32_f16(false, a, false, b1, (short)0, c1, false, false);
    c2 = __builtin_amdgcn_wmma_f32_16x16x32_f16(false, a, false, b2, (short)0, c2, false, false);
    c3 = __builtin_amdgcn_wmma_f32_16x16x32_f16(false, a, false, b3, (short)0, c3, false, false);
  }
  v8f acc[4] = {c0, c1, c2, c3};
#pragma unroll
  for (int t = 0; t < 4; ++t) {
    int oc = o0 + t * 16 + lo;
    float bv = bias ? bias[oc] : 0.f;
#pragma unroll
    for (int r = 0; r < 8; ++r) {
      int m = m0 + hi * 8 + r;
      float v = acc[t][r] + bv;
      if (relu) v = v > 0.f ? v : 0.f;
      if (mode == 0)      outH[(size_t)m * O + oc] = (_Float16)v;
      else if (mode == 1) outF[(size_t)m * O + oc] = v;
      else                atomicMaxFloat(&outF[(size_t)(m >> 11) * O + oc], v);
    }
  }
}

// ---------------------------------------------------------------------------
// local_maxpool: P[m,c] = max_j X[(b,idx[m,j]),c]   (block = C threads)
// ---------------------------------------------------------------------------
__global__ void maxpool_k(const _Float16* __restrict__ X,
                          const int* __restrict__ idx,
                          _Float16* __restrict__ P, int C) {
  int m = blockIdx.x;
  int b = m >> 11;
  int c = threadIdx.x;
  const int* ip = idx + (size_t)m * KNN;
  float mx = -3.4e38f;
#pragma unroll
  for (int j = 0; j < KNN; ++j) {
    int r = ip[j];
    float v = (float)X[((size_t)b * NPTS + r) * C + c];
    mx = fmaxf(mx, v);
  }
  P[(size_t)m * C + c] = (_Float16)mx;
}

__global__ void ginit_k(float* g) {
  g[blockIdx.x * 256 + threadIdx.x] = -__builtin_inff();
}

__global__ void f2h_k(const float* __restrict__ s, _Float16* __restrict__ d, int n) {
  int i = blockIdx.x * 256 + threadIdx.x;
  if (i < n) d[i] = (_Float16)s[i];
}

// ---------------------------------------------------------------------------
extern "C" void kernel_launch(void* const* d_in, const int* in_sizes, int n_in,
                              void* d_out, int out_size, void* d_ws, size_t ws_size,
                              hipStream_t stream) {
  const float* pts    = (const float*)d_in[0];
  const float* W_m1a  = (const float*)d_in[1];  const float* b_m1a = (const float*)d_in[2];
  const float* W_m1b  = (const float*)d_in[3];  const float* b_m1b = (const float*)d_in[4];
  const float* W_m1c  = (const float*)d_in[5];  const float* b_m1c = (const float*)d_in[6];
  const float* W_lin1 = (const float*)d_in[7];  const float* b_lin1 = (const float*)d_in[8];
  const float* W_c1   = (const float*)d_in[9];  const float* b_c1  = (const float*)d_in[10];
  const float* W_lin2 = (const float*)d_in[11]; const float* b_lin2 = (const float*)d_in[12];
  const float* W_c2   = (const float*)d_in[13]; const float* b_c2  = (const float*)d_in[14];
  const float* W_m2a  = (const float*)d_in[15]; const float* b_m2a = (const float*)d_in[16];
  const float* W_m2b  = (const float*)d_in[17]; const float* b_m2b = (const float*)d_in[18];
  const float* W_emb  = (const float*)d_in[19];
  float* out = (float*)d_out;

  char* ws = (char*)d_ws;
  size_t cur = 0;
  auto alloc = [&](size_t bytes) -> char* {
    char* p = ws + cur;
    cur += (bytes + 255) & ~(size_t)255;
    return p;
  };
  int*      idx   = (int*)alloc((size_t)BN * KNN * 4);
  _Float16* bufA  = (_Float16*)alloc((size_t)BN * 128 * 2);
  _Float16* bufB  = (_Float16*)alloc((size_t)BN * 128 * 2);
  _Float16* wm1a  = (_Float16*)alloc(64 * 32 * 2);
  _Float16* wm1b  = (_Float16*)alloc(64 * 64 * 2);
  _Float16* wm1c  = (_Float16*)alloc(64 * 64 * 2);
  _Float16* wlin1 = (_Float16*)alloc(64 * 64 * 2);
  _Float16* wc1   = (_Float16*)alloc(128 * 64 * 2);
  _Float16* wlin2 = (_Float16*)alloc(128 * 128 * 2);
  _Float16* wc2   = (_Float16*)alloc(1024 * 128 * 2);
  _Float16* wm2a  = (_Float16*)alloc((size_t)512 * 1024 * 2);
  _Float16* wm2b  = (_Float16*)alloc((size_t)512 * 512 * 2);
  _Float16* wemb  = (_Float16*)alloc((size_t)128 * 512 * 2);
  float*    gmax  = (float*)alloc(BATCH * 1024 * 4);
  _Float16* gmaxh = (_Float16*)alloc(BATCH * 1024 * 2);
  _Float16* y1    = (_Float16*)alloc(BATCH * 512 * 2);
  _Float16* y2    = (_Float16*)alloc(BATCH * 512 * 2);

  // 1) KNN + cov features (K padded to 32)
  knn_k<<<BATCH * (NPTS / 256), 256, 0, stream>>>(pts, idx);
  cov_k<<<BN / 256, 256, 0, stream>>>(pts, idx, bufA);

  // 2) weight conversion (pad m1a K 12->32)
  auto wcv = [&](const float* s, _Float16* d, int O, int Ks, int Kp) {
    wcvt_k<<<(O * Kp + 255) / 256, 256, 0, stream>>>(s, d, O, Ks, Kp);
  };
  wcv(W_m1a, wm1a, 64, 12, 32);
  wcv(W_m1b, wm1b, 64, 64, 64);
  wcv(W_m1c, wm1c, 64, 64, 64);
  wcv(W_lin1, wlin1, 64, 64, 64);
  wcv(W_c1, wc1, 128, 64, 64);
  wcv(W_lin2, wlin2, 128, 128, 128);
  wcv(W_c2, wc2, 1024, 128, 128);
  wcv(W_m2a, wm2a, 512, 1024, 1024);
  wcv(W_m2b, wm2b, 512, 512, 512);
  wcv(W_emb, wemb, 128, 512, 512);

  // 3) per-point MLP chain (WMMA GEMMs, (BN,C) f16 row-major, ping-pong)
  dim3 g64(BN / 16, 64 / 64), g128(BN / 16, 128 / 64), g1024(BN / 16, 1024 / 64);
  gemm_k<<<g64, 32, 0, stream>>>(bufA, wm1a, b_m1a, bufB, nullptr, 32, 64, 1, 0);
  gemm_k<<<g64, 32, 0, stream>>>(bufB, wm1b, b_m1b, bufA, nullptr, 64, 64, 1, 0);
  gemm_k<<<g64, 32, 0, stream>>>(bufA, wm1c, b_m1c, bufB, nullptr, 64, 64, 1, 0);
  maxpool_k<<<BN, 64, 0, stream>>>(bufB, idx, bufA, 64);
  gemm_k<<<g64, 32, 0, stream>>>(bufA, wlin1, b_lin1, bufB, nullptr, 64, 64, 0, 0);
  gemm_k<<<g128, 32, 0, stream>>>(bufB, wc1, b_c1, bufA, nullptr, 64, 128, 1, 0);
  maxpool_k<<<BN, 128, 0, stream>>>(bufA, idx, bufB, 128);
  gemm_k<<<g128, 32, 0, stream>>>(bufB, wlin2, b_lin2, bufA, nullptr, 128, 128, 0, 0);

  // 4) 128->1024 conv fused with global max over points (atomic f32 max)
  ginit_k<<<BATCH * 1024 / 256, 256, 0, stream>>>(gmax);
  gemm_k<<<g1024, 32, 0, stream>>>(bufA, wc2, b_c2, nullptr, gmax, 128, 1024, 0, 2);
  f2h_k<<<BATCH * 1024 / 256, 256, 0, stream>>>(gmax, gmaxh, BATCH * 1024);

  // 5) head MLP (M = 16 batch rows)
  gemm_k<<<dim3(1, 512 / 64), 32, 0, stream>>>(gmaxh, wm2a, b_m2a, y1, nullptr, 1024, 512, 1, 0);
  gemm_k<<<dim3(1, 512 / 64), 32, 0, stream>>>(y1, wm2b, b_m2b, y2, nullptr, 512, 512, 0, 0);
  gemm_k<<<dim3(1, 128 / 64), 32, 0, stream>>>(y2, wemb, nullptr, nullptr, out, 512, 128, 0, 1);
}
